// HeteroTransportCell_29076928594593
// MI455X (gfx1250) — matrix operand order repaired and checked by reading, not compiled, not run on hardware
//
#include <hip/hip_runtime.h>

// ---------------------------------------------------------------------------
// Types / helpers
// ---------------------------------------------------------------------------
typedef _Float16 half_t;
typedef __attribute__((ext_vector_type(16))) _Float16 v16h;
typedef __attribute__((ext_vector_type(8)))  _Float16 v8h;
typedef __attribute__((ext_vector_type(8)))  float    v8f;

#define DEV __device__ __forceinline__

DEV void wave_lds_fence() { asm volatile("s_wait_dscnt 0" ::: "memory"); }

DEV v8f wmma_f16(v16h a, v16h b, v8f c) {
  return __builtin_amdgcn_wmma_f32_16x16x32_f16(false, a, false, b, (short)0, c,
                                                false, false);
}

// A operand: 16x32 f16 tile, row m = lane%16, two contiguous 8-half chunks.
DEV v16h load_A(const half_t* X, int ld, int lane, int kb) {
  int m = lane & 15;
  int k0 = kb + ((lane & 16) ? 16 : 0);
  const half_t* p = X + m * ld + k0;
  v8h lo = *(const v8h*)(p);
  v8h hi = *(const v8h*)(p + 8);
  v16h a;
#pragma unroll
  for (int t = 0; t < 8; ++t) { a[t] = lo[t]; a[8 + t] = hi[t]; }
  return a;
}

// B operand: W row-major [N][K]; lane holds col n = nb+lane%16,
// 16 contiguous halves starting at kb + (lane>=16 ? 16 : 0).
DEV v16h load_B(const half_t* W, int ldk, int lane, int kb, int nb) {
  int n = nb + (lane & 15);
  int k0 = kb + ((lane & 16) ? 16 : 0);
  return *(const v16h*)(W + (size_t)n * ldk + k0);
}

template <int KT, int NT>
DEV void gemm(const half_t* A, int lda, const half_t* W, int ldw, int lane,
              v8f* acc) {
#pragma unroll
  for (int kt = 0; kt < KT; ++kt) {
    v16h a = load_A(A, lda, lane, kt * 32);
#pragma unroll
    for (int nt = 0; nt < NT; ++nt) {
      v16h b = load_B(W, ldw, lane, kt * 32, nt * 16);
      acc[nt] = wmma_f16(a, b, acc[nt]);
    }
  }
}

template <int NT>
DEV void zero_acc(v8f* acc) {
#pragma unroll
  for (int nt = 0; nt < NT; ++nt)
#pragma unroll
    for (int e = 0; e < 8; ++e) acc[nt][e] = 0.0f;
}

// store acc (+bias, optional relu) as halves into LDS [16][ld]
template <int NT>
DEV void store_acts(half_t* X, int ld, int lane, const v8f* acc,
                    const float* bias, bool relu) {
  int n0 = lane & 15;
  int mb = (lane & 16) ? 8 : 0;
#pragma unroll
  for (int nt = 0; nt < NT; ++nt) {
    int n = n0 + nt * 16;
    float b = bias[n];
#pragma unroll
    for (int e = 0; e < 8; ++e) {
      float v = acc[nt][e] + b;
      if (relu) v = fmaxf(v, 0.0f);
      X[(mb + e) * ld + n] = (half_t)v;
    }
  }
}

DEV float sigmoidf_(float x) { return 1.0f / (1.0f + __expf(-x)); }
DEV float softplusf_(float x) { return (x > 20.0f) ? x : log1pf(__expf(x)); }

// order-preserving float<->uint mapping for atomicMax on floats
DEV unsigned fkey(float f) {
  unsigned u = __float_as_uint(f);
  return (u & 0x80000000u) ? ~u : (u | 0x80000000u);
}
DEV float funkey(unsigned u) {
  return (u & 0x80000000u) ? __uint_as_float(u & 0x7fffffffu)
                           : __uint_as_float(~u);
}

// ---------------------------------------------------------------------------
// Weight convert/pad f32 -> f16, dst row-major [rows][dcols], zero pad cols.
// ---------------------------------------------------------------------------
__global__ void k_cvt_pad(half_t* dst, const float* src, int rows, int scols,
                          int dcols) {
  int idx = blockIdx.x * blockDim.x + threadIdx.x;
  int total = rows * dcols;
  if (idx >= total) return;
  int r = idx / dcols, c = idx - r * dcols;
  float v = (c < scols) ? src[r * scols + c] : 0.0f;
  dst[idx] = (half_t)v;
}

// ---------------------------------------------------------------------------
// Fused StaticDynamicEdgeMP: per 16-edge tile (1 wave), 8 tiles/block.
// ---------------------------------------------------------------------------
#define SD_TILES 8
__global__ __launch_bounds__(256, 1) void k_sd_edge(
    int E, const int* __restrict__ ei, const float* __restrict__ ea,
    const float* __restrict__ xs, const float* __restrict__ h,
    const half_t* __restrict__ gW1,  // [128][32] emb fc1 (K padded 19->32)
    const half_t* __restrict__ gW2,  // [128][128] emb fc2 (global B)
    const half_t* __restrict__ gG1,  // [128][128] gate fc1 (global B)
    const half_t* __restrict__ gBG,  // [16][128] row0=base.w row1=gate2.w
    const half_t* __restrict__ gP1,  // [128][64] payload fc1
    const half_t* __restrict__ gP2,  // [64][128] payload fc2
    const float* __restrict__ b1, const float* __restrict__ b2,
    const float* __restrict__ base_b, const float* __restrict__ bg1,
    const float* __restrict__ gate2_b, const float* __restrict__ bp1,
    const float* __restrict__ bp2, float* __restrict__ M) {
  __shared__ half_t sW1[128 * 32];
  __shared__ half_t sBG[16 * 128];
  __shared__ half_t sP1[128 * 64];
  __shared__ half_t sP2[64 * 128];
  __shared__ float sB1[128], sB2[128], sBg1[128], sBp1[128], sBp2[64];
  __shared__ half_t sX[SD_TILES][16 * 128];
  __shared__ half_t sY[SD_TILES][16 * 128];
  __shared__ float sScale[SD_TILES][16];
  __shared__ int sDst[SD_TILES][16];

  int tid = threadIdx.x;
  for (int k = tid; k < 128 * 32; k += 256) sW1[k] = gW1[k];
  for (int k = tid; k < 16 * 128; k += 256) sBG[k] = gBG[k];
  for (int k = tid; k < 128 * 64; k += 256) sP1[k] = gP1[k];
  for (int k = tid; k < 64 * 128; k += 256) sP2[k] = gP2[k];
  for (int k = tid; k < 128; k += 256) {
    sB1[k] = b1[k]; sB2[k] = b2[k]; sBg1[k] = bg1[k]; sBp1[k] = bp1[k];
  }
  for (int k = tid; k < 64; k += 256) sBp2[k] = bp2[k];
  __syncthreads();

  int wave = tid >> 5, lane = tid & 31;
  long tile0 = ((long)blockIdx.x * SD_TILES + wave) * 16;
  if (tile0 >= E) return;
  half_t* X = sX[wave];
  half_t* Y = sY[wave];

  int m = lane & 15, p = lane >> 4;
  {
    long e = tile0 + m;
    bool valid = e < (long)E;
    long ec = valid ? e : (long)(E - 1);
    int j = ei[ec];
    int i = ei[(size_t)E + ec];
    if (p == 0) sDst[wave][m] = valid ? i : -1;
    // hcat = [h[j] | h[i]]  (lane p=0 -> src, p=1 -> dst)
    const float* hr = h + (size_t)(p ? i : j) * 64;
    half_t* dy = Y + m * 128 + p * 64;
#pragma unroll
    for (int c = 0; c < 64; c += 4) {
      float4 f = *(const float4*)(hr + c);
      dy[c] = (half_t)f.x; dy[c + 1] = (half_t)f.y;
      dy[c + 2] = (half_t)f.z; dy[c + 3] = (half_t)f.w;
    }
    if (p == 0) {
      half_t* dx = X + m * 128;
      const float* eap = ea + (size_t)ec * 3;
      dx[0] = (half_t)eap[0]; dx[1] = (half_t)eap[1]; dx[2] = (half_t)eap[2];
      const float* xj = xs + (size_t)j * 8;
      const float* xi = xs + (size_t)i * 8;
#pragma unroll
      for (int c = 0; c < 8; ++c) {
        dx[3 + c] = (half_t)xj[c];
        dx[11 + c] = (half_t)xi[c];
      }
#pragma unroll
      for (int c = 19; c < 32; ++c) dx[c] = (half_t)0.0f;
    }
  }
  wave_lds_fence();

  v8f acc[8];
  // emb fc1: relu(Xin[:, :32] @ W1^T + b1) -> X
  zero_acc<8>(acc);
  gemm<1, 8>(X, 128, sW1, 32, lane, acc);
  store_acts<8>(X, 128, lane, acc, sB1, true);
  wave_lds_fence();
  // emb fc2: u = X @ W2^T + b2 -> X
  zero_acc<8>(acc);
  gemm<4, 8>(X, 128, gW2, 128, lane, acc);
  store_acts<8>(X, 128, lane, acc, sB2, false);
  wave_lds_fence();
  // base = softplus(u @ BG^T [col0] + base_b)
  {
    v8f a0; zero_acc<1>(&a0);
    gemm<4, 1>(X, 128, sBG, 128, lane, &a0);
    if ((lane & 15) == 0) {
      int mb = (lane & 16) ? 8 : 0;
      float bb = base_b[0];
#pragma unroll
      for (int e = 0; e < 8; ++e) sScale[wave][mb + e] = softplusf_(a0[e] + bb);
    }
  }
  wave_lds_fence();
  // gate fc1: relu(Y @ G1^T + bg1) -> X
  zero_acc<8>(acc);
  gemm<4, 8>(Y, 128, gG1, 128, lane, acc);
  store_acts<8>(X, 128, lane, acc, sBg1, true);
  wave_lds_fence();
  // gate fc2 [col1] -> sigmoid, fold into scale
  {
    v8f a0; zero_acc<1>(&a0);
    gemm<4, 1>(X, 128, sBG, 128, lane, &a0);
    if ((lane & 15) == 1) {
      int mb = (lane & 16) ? 8 : 0;
      float bb = gate2_b[0];
#pragma unroll
      for (int e = 0; e < 8; ++e)
        sScale[wave][mb + e] *= sigmoidf_(a0[e] + bb);
    }
  }
  wave_lds_fence();
  // payload fc1: relu(h_src @ P1^T + bp1) -> X   (h_src = Y[:, :64])
  zero_acc<8>(acc);
  gemm<2, 8>(Y, 128, sP1, 64, lane, acc);
  store_acts<8>(X, 128, lane, acc, sBp1, true);
  wave_lds_fence();
  // payload fc2 + scaled scatter-add
  zero_acc<4>(acc);
  gemm<4, 4>(X, 128, sP2, 128, lane, acc);
  {
    int n0 = lane & 15, mb = (lane & 16) ? 8 : 0;
#pragma unroll
    for (int nt = 0; nt < 4; ++nt) {
      int n = n0 + nt * 16;
      float bb = sBp2[n];
#pragma unroll
      for (int e = 0; e < 8; ++e) {
        int mm = mb + e;
        int dst = sDst[wave][mm];
        if (dst >= 0)
          atomicAdd(&M[(size_t)dst * 64 + n],
                    sScale[wave][mm] * (acc[nt][e] + bb));
      }
    }
  }
}

// ---------------------------------------------------------------------------
// Node projection: out_f16[n,64] = in[n,64] @ W^T + b  (GAT lin_l / lin_r)
// ---------------------------------------------------------------------------
__global__ __launch_bounds__(256, 1) void k_lin64(int N,
                                                  const float* __restrict__ in,
                                                  const half_t* __restrict__ gW,
                                                  const float* __restrict__ bias,
                                                  half_t* __restrict__ out) {
  __shared__ half_t sW[64 * 64];
  __shared__ float sB[64];
  __shared__ half_t sXt[8][16 * 64];
  int tid = threadIdx.x;
  for (int k = tid; k < 64 * 64; k += 256) sW[k] = gW[k];
  for (int k = tid; k < 64; k += 256) sB[k] = bias[k];
  __syncthreads();
  int wave = tid >> 5, lane = tid & 31;
  long base = ((long)blockIdx.x * 8 + wave) * 16;
  if (base >= N) return;
  int m = lane & 15, p = lane >> 4;
  long node = base + m;
  long nc = node < N ? node : (long)N - 1;
  {
    const float* src = in + (size_t)nc * 64 + p * 32;
    half_t* d = sXt[wave] + m * 64 + p * 32;
#pragma unroll
    for (int c = 0; c < 32; c += 4) {
      float4 f = *(const float4*)(src + c);
      d[c] = (half_t)f.x; d[c + 1] = (half_t)f.y;
      d[c + 2] = (half_t)f.z; d[c + 3] = (half_t)f.w;
    }
  }
  wave_lds_fence();
  v8f acc[4];
  zero_acc<4>(acc);
  gemm<2, 4>(sXt[wave], 64, sW, 64, lane, acc);
  int n0 = lane & 15, mb = (lane & 16) ? 8 : 0;
#pragma unroll
  for (int nt = 0; nt < 4; ++nt) {
    int n = n0 + nt * 16;
    float bb = sB[n];
#pragma unroll
    for (int e = 0; e < 8; ++e) {
      long nd = base + mb + e;
      if (nd < N) out[(size_t)nd * 64 + n] = (half_t)(acc[nt][e] + bb);
    }
  }
}

// ---------------------------------------------------------------------------
// GATv2 edge passes
// ---------------------------------------------------------------------------
__global__ void k_gat_logit(int E, const int* __restrict__ ei,
                            const half_t* __restrict__ xl,
                            const half_t* __restrict__ xr,
                            const float* __restrict__ att,
                            float* __restrict__ logit,
                            unsigned* __restrict__ mx) {
  int e = blockIdx.x * blockDim.x + threadIdx.x;
  if (e >= E) return;
  int j = ei[e], i = ei[(size_t)E + e];
  const half_t* a = xl + (size_t)j * 64;
  const half_t* b = xr + (size_t)i * 64;
#pragma unroll
  for (int hh = 0; hh < 4; ++hh) {
    float lg = 0.0f;
#pragma unroll
    for (int c = 0; c < 16; ++c) {
      float s = (float)a[hh * 16 + c] + (float)b[hh * 16 + c];
      s = (s > 0.0f) ? s : 0.2f * s;  // leaky_relu(0.2)
      lg += s * att[hh * 16 + c];
    }
    logit[(size_t)e * 4 + hh] = lg;
    atomicMax(&mx[(size_t)i * 4 + hh], fkey(lg));
  }
}

__global__ void k_gat_den(int E, const int* __restrict__ ei,
                          const unsigned* __restrict__ mx,
                          float* __restrict__ logit, float* __restrict__ den) {
  int e = blockIdx.x * blockDim.x + threadIdx.x;
  if (e >= E) return;
  int i = ei[(size_t)E + e];
#pragma unroll
  for (int hh = 0; hh < 4; ++hh) {
    float mv = funkey(mx[(size_t)i * 4 + hh]);
    float ex = __expf(logit[(size_t)e * 4 + hh] - mv);
    logit[(size_t)e * 4 + hh] = ex;
    atomicAdd(&den[(size_t)i * 4 + hh], ex);
  }
}

__global__ void k_gat_scatter(int E, const int* __restrict__ ei,
                              const half_t* __restrict__ xl,
                              const float* __restrict__ ex,
                              const float* __restrict__ den,
                              float* __restrict__ gacc) {
  int e = blockIdx.x * blockDim.x + threadIdx.x;
  if (e >= E) return;
  int j = ei[e], i = ei[(size_t)E + e];
  const half_t* a = xl + (size_t)j * 64;
#pragma unroll
  for (int hh = 0; hh < 4; ++hh) {
    float alpha =
        ex[(size_t)e * 4 + hh] / (den[(size_t)i * 4 + hh] + 1e-16f);
#pragma unroll
    for (int c = 0; c < 16; ++c)
      atomicAdd(&gacc[(size_t)i * 64 + hh * 16 + c],
                alpha * (float)a[hh * 16 + c]);
  }
}

// ---------------------------------------------------------------------------
// GAT node tail: out = gacc + h2 @ res^T + bias ; M2 += FFN(out)
// ---------------------------------------------------------------------------
__global__ __launch_bounds__(256, 1) void k_gat_node(
    int N, const float* __restrict__ gacc, const float* __restrict__ h,
    const half_t* __restrict__ gWres, const float* __restrict__ gbias,
    const half_t* __restrict__ gF1, const float* __restrict__ bf1,
    const half_t* __restrict__ gF2, const float* __restrict__ bf2,
    float* __restrict__ M2) {
  __shared__ half_t sWr[64 * 64];
  __shared__ half_t sF1[128 * 64];
  __shared__ half_t sF2[64 * 128];
  __shared__ float sB[64], sB1[128], sB2[64];
  __shared__ half_t sXh[8][16 * 64];
  __shared__ half_t sXo[8][16 * 128];
  int tid = threadIdx.x;
  for (int k = tid; k < 64 * 64; k += 256) sWr[k] = gWres[k];
  for (int k = tid; k < 128 * 64; k += 256) sF1[k] = gF1[k];
  for (int k = tid; k < 64 * 128; k += 256) sF2[k] = gF2[k];
  for (int k = tid; k < 128; k += 256) sB1[k] = bf1[k];
  for (int k = tid; k < 64; k += 256) { sB[k] = gbias[k]; sB2[k] = bf2[k]; }
  __syncthreads();
  int wave = tid >> 5, lane = tid & 31;
  long base = ((long)blockIdx.x * 8 + wave) * 16;
  if (base >= N) return;
  int m = lane & 15, p = lane >> 4;
  long nc = base + m;
  if (nc >= N) nc = N - 1;
  {
    const float* src = h + (size_t)nc * 64 + p * 32;
    half_t* d = sXh[wave] + m * 64 + p * 32;
#pragma unroll
    for (int c = 0; c < 32; c += 4) {
      float4 f = *(const float4*)(src + c);
      d[c] = (half_t)f.x; d[c + 1] = (half_t)f.y;
      d[c + 2] = (half_t)f.z; d[c + 3] = (half_t)f.w;
    }
  }
  wave_lds_fence();
  v8f acc[8];
  zero_acc<4>(acc);
  gemm<2, 4>(sXh[wave], 64, sWr, 64, lane, acc);
  {
    int n0 = lane & 15, mb = (lane & 16) ? 8 : 0;
#pragma unroll
    for (int nt = 0; nt < 4; ++nt) {
      int n = n0 + nt * 16;
#pragma unroll
      for (int e = 0; e < 8; ++e) {
        int mm = mb + e;
        long nd = base + mm;
        long ndc = nd < N ? nd : (long)N - 1;
        float o = acc[nt][e] + sB[n] + gacc[(size_t)ndc * 64 + n];
        sXo[wave][mm * 128 + n] = (half_t)o;
      }
    }
  }
  wave_lds_fence();
  zero_acc<8>(acc);
  gemm<2, 8>(sXo[wave], 128, sF1, 64, lane, acc);
  store_acts<8>(sXo[wave], 128, lane, acc, sB1, true);
  wave_lds_fence();
  zero_acc<4>(acc);
  gemm<4, 4>(sXo[wave], 128, sF2, 128, lane, acc);
  {
    int n0 = lane & 15, mb = (lane & 16) ? 8 : 0;
#pragma unroll
    for (int nt = 0; nt < 4; ++nt) {
      int n = n0 + nt * 16;
      float bb = sB2[n];
#pragma unroll
      for (int e = 0; e < 8; ++e) {
        long nd = base + mb + e;
        if (nd < N) M2[(size_t)nd * 64 + n] += acc[nt][e] + bb;
      }
    }
  }
}

// ---------------------------------------------------------------------------
// Node update: LN(M), LN(dyn_proj(xdyn)), GRU (WMMA), LN -> out
// ---------------------------------------------------------------------------
__global__ __launch_bounds__(256, 1) void k_node_update(
    int N, const float* __restrict__ Mi, const float* __restrict__ xdyn,
    const float* __restrict__ h, const half_t* __restrict__ gWih,
    const half_t* __restrict__ gWhh, const float* __restrict__ b_ih,
    const float* __restrict__ b_hh, const float* __restrict__ dynW,
    const float* __restrict__ dynB, const float* __restrict__ mg,
    const float* __restrict__ mb_, const float* __restrict__ dg,
    const float* __restrict__ db_, const float* __restrict__ hg,
    const float* __restrict__ hb_, float* __restrict__ out) {
  __shared__ half_t sWhh[192 * 64];
  __shared__ float sBih[192], sBhh[192];
  __shared__ half_t sXh[8][16 * 64];
  __shared__ half_t sXx[8][16 * 128];
  __shared__ float sHn[8][16 * 64];
  int tid = threadIdx.x;
  for (int k = tid; k < 192 * 64; k += 256) sWhh[k] = gWhh[k];
  for (int k = tid; k < 192; k += 256) { sBih[k] = b_ih[k]; sBhh[k] = b_hh[k]; }
  __syncthreads();
  int wave = tid >> 5, lane = tid & 31;
  long base = ((long)blockIdx.x * 8 + wave) * 16;
  if (base >= N) return;
  int m = lane & 15, p = lane >> 4;
  long node = base + m;
  long nc = node < N ? node : (long)N - 1;
  // stage h tile as halves
  {
    const float* src = h + (size_t)nc * 64 + p * 32;
    half_t* d = sXh[wave] + m * 64 + p * 32;
#pragma unroll
    for (int c = 0; c < 32; c += 4) {
      float4 f = *(const float4*)(src + c);
      d[c] = (half_t)f.x; d[c + 1] = (half_t)f.y;
      d[c + 2] = (half_t)f.z; d[c + 3] = (half_t)f.w;
    }
  }
  float mv[32];
  // LN(M) -> sXx[:, 0:64]
  {
    const float* src = Mi + (size_t)nc * 64 + p * 32;
    float s = 0.0f, s2 = 0.0f;
#pragma unroll
    for (int c = 0; c < 32; ++c) { mv[c] = src[c]; s += mv[c]; s2 += mv[c] * mv[c]; }
    s += __shfl_xor(s, 16);
    s2 += __shfl_xor(s2, 16);
    float mu = s * (1.0f / 64.0f);
    float rs = rsqrtf(s2 * (1.0f / 64.0f) - mu * mu + 1e-5f);
    half_t* d = sXx[wave] + m * 128;
#pragma unroll
    for (int c = 0; c < 32; ++c) {
      int cc = p * 32 + c;
      d[cc] = (half_t)((mv[c] - mu) * rs * mg[cc] + mb_[cc]);
    }
  }
  // LN(dyn_proj(xdyn)) -> sXx[:, 64:128]
  {
    const float* xp = xdyn + (size_t)nc * 4;
    float x0 = xp[0], x1 = xp[1], x2 = xp[2], x3 = xp[3];
    float s = 0.0f, s2 = 0.0f;
#pragma unroll
    for (int c = 0; c < 32; ++c) {
      int cc = p * 32 + c;
      float v = dynB[cc] + x0 * dynW[cc * 4 + 0] + x1 * dynW[cc * 4 + 1] +
                x2 * dynW[cc * 4 + 2] + x3 * dynW[cc * 4 + 3];
      mv[c] = v; s += v; s2 += v * v;
    }
    s += __shfl_xor(s, 16);
    s2 += __shfl_xor(s2, 16);
    float mu = s * (1.0f / 64.0f);
    float rs = rsqrtf(s2 * (1.0f / 64.0f) - mu * mu + 1e-5f);
    half_t* d = sXx[wave] + m * 128 + 64;
#pragma unroll
    for (int c = 0; c < 32; ++c) {
      int cc = p * 32 + c;
      d[cc] = (half_t)((mv[c] - mu) * rs * dg[cc] + db_[cc]);
    }
  }
  wave_lds_fence();
  v8f gh[12]; zero_acc<12>(gh);
  gemm<2, 12>(sXh[wave], 64, sWhh, 64, lane, gh);
  v8f gi[12]; zero_acc<12>(gi);
  gemm<4, 12>(sXx[wave], 128, gWih, 128, lane, gi);
  {
    int n0 = lane & 15, mb2 = (lane & 16) ? 8 : 0;
#pragma unroll
    for (int nt = 0; nt < 4; ++nt) {
      int n = n0 + nt * 16;
#pragma unroll
      for (int e = 0; e < 8; ++e) {
        int mm = mb2 + e;
        float r = sigmoidf_(gi[nt][e] + sBih[n] + gh[nt][e] + sBhh[n]);
        float z = sigmoidf_(gi[nt + 4][e] + sBih[64 + n] + gh[nt + 4][e] +
                            sBhh[64 + n]);
        float nn = tanhf(gi[nt + 8][e] + sBih[128 + n] +
                         r * (gh[nt + 8][e] + sBhh[128 + n]));
        float hp = (float)sXh[wave][mm * 64 + n];
        sHn[wave][mm * 64 + n] = (1.0f - z) * nn + z * hp;
      }
    }
  }
  wave_lds_fence();
  // final LN(h_new) -> out
  {
    const float* src = sHn[wave] + m * 64 + p * 32;
    float s = 0.0f, s2 = 0.0f;
#pragma unroll
    for (int c = 0; c < 32; ++c) { mv[c] = src[c]; s += mv[c]; s2 += mv[c] * mv[c]; }
    s += __shfl_xor(s, 16);
    s2 += __shfl_xor(s2, 16);
    float mu = s * (1.0f / 64.0f);
    float rs = rsqrtf(s2 * (1.0f / 64.0f) - mu * mu + 1e-5f);
    if (node < N) {
      float* d = out + (size_t)node * 64 + p * 32;
#pragma unroll
      for (int c = 0; c < 32; ++c) {
        int cc = p * 32 + c;
        d[c] = (mv[c] - mu) * rs * hg[cc] + hb_[cc];
      }
    }
  }
}

// ---------------------------------------------------------------------------
// Host launcher
// ---------------------------------------------------------------------------
extern "C" void kernel_launch(void* const* d_in, const int* in_sizes, int n_in,
                              void* d_out, int out_size, void* d_ws,
                              size_t ws_size, hipStream_t stream) {
  // Leaf indices (setup_inputs() dict insertion order, nested)
  enum {
    I_MP1 = 0, I_MP2 = 14, I_GAT = 28, I_NT1 = 39, I_NT2 = 51,
    I_H1 = 63, I_H2 = 64, I_XD1 = 65, I_XD2 = 66, I_XS1 = 67, I_XS2 = 68,
    I_EA1 = 69, I_EA2 = 70, I_EI1 = 71, I_EI2 = 72, I_EIX = 73
  };
  (void)n_in; (void)out_size; (void)ws_size;
  const long BN1 = in_sizes[I_H1] / 64;
  const long BN2 = in_sizes[I_H2] / 64;
  const int BE1 = in_sizes[I_EI1] / 2;
  const int BE2 = in_sizes[I_EI2] / 2;
  const int BE12 = in_sizes[I_EIX] / 2;

  // f16 weight block layout (element offsets)
  constexpr size_t MP_W1 = 0;
  constexpr size_t MP_W2 = MP_W1 + 128 * 32;
  constexpr size_t MP_G1 = MP_W2 + 128 * 128;
  constexpr size_t MP_BG = MP_G1 + 128 * 128;
  constexpr size_t MP_P1 = MP_BG + 16 * 128;
  constexpr size_t MP_P2 = MP_P1 + 128 * 64;
  constexpr size_t MP_TOT = MP_P2 + 64 * 128;
  constexpr size_t G_LL = 2 * MP_TOT;
  constexpr size_t G_LR = G_LL + 64 * 64;
  constexpr size_t G_RES = G_LR + 64 * 64;
  constexpr size_t G_F1 = G_RES + 64 * 64;
  constexpr size_t G_F2 = G_F1 + 128 * 64;
  constexpr size_t NT_IH1 = G_F2 + 64 * 128;
  constexpr size_t NT_HH1 = NT_IH1 + 192 * 128;
  constexpr size_t NT_IH2 = NT_HH1 + 192 * 64;
  constexpr size_t NT_HH2 = NT_IH2 + 192 * 128;
  constexpr size_t WF16_TOT = NT_HH2 + 192 * 64;

  char* ws = (char*)d_ws;
  size_t off = 0;
  auto take = [&](size_t bytes) {
    size_t o = off;
    off += (bytes + 255) & ~(size_t)255;
    return o;
  };
  half_t* WB = (half_t*)(ws + take(WF16_TOT * sizeof(half_t)));
  size_t oM1 = take((size_t)BN1 * 64 * 4);
  float* M1 = (float*)(ws + oM1);
  float* M2 = (float*)(ws + take((size_t)BN2 * 64 * 4));
  unsigned* MAXB = (unsigned*)(ws + take((size_t)BN2 * 4 * 4));
  float* DEN = (float*)(ws + take((size_t)BN2 * 4 * 4));
  float* GACC = (float*)(ws + take((size_t)BN2 * 64 * 4));
  size_t zeroEnd = off;
  half_t* XL = (half_t*)(ws + take((size_t)BN1 * 64 * 2));
  half_t* XR = (half_t*)(ws + take((size_t)BN2 * 64 * 2));
  float* LG = (float*)(ws + take((size_t)BE12 * 4 * 4));

  hipMemsetAsync(ws + oM1, 0, zeroEnd - oM1, stream);

  auto cvt = [&](size_t dstOff, int leaf, int rows, int scols, int dcols) {
    int total = rows * dcols;
    k_cvt_pad<<<dim3((total + 255) / 256), 256, 0, stream>>>(
        WB + dstOff, (const float*)d_in[leaf], rows, scols, dcols);
  };
  for (int g = 0; g < 2; ++g) {
    int b = g ? I_MP2 : I_MP1;
    size_t o = g ? MP_TOT : 0;
    cvt(o + MP_W1, b + 0, 128, 19, 32);       // emb.fc1.w  (pad K 19->32)
    cvt(o + MP_W2, b + 2, 128, 128, 128);     // emb.fc2.w
    cvt(o + MP_G1, b + 6, 128, 128, 128);     // gate.fc1.w
    cvt(o + MP_BG, b + 4, 16, 0, 128);        // zero 16x128
    cvt(o + MP_BG, b + 4, 1, 128, 128);       // row0 = base.w
    cvt(o + MP_BG + 128, b + 8, 1, 128, 128); // row1 = gate.fc2.w
    cvt(o + MP_P1, b + 10, 128, 64, 64);      // payload.fc1.w
    cvt(o + MP_P2, b + 12, 64, 128, 128);     // payload.fc2.w
  }
  cvt(G_LL, I_GAT + 0, 64, 64, 64);
  cvt(G_LR, I_GAT + 2, 64, 64, 64);
  cvt(G_RES, I_GAT + 5, 64, 64, 64);
  cvt(G_F1, I_GAT + 7, 128, 64, 64);
  cvt(G_F2, I_GAT + 9, 64, 128, 128);
  cvt(NT_IH1, I_NT1 + 2, 192, 128, 128);
  cvt(NT_HH1, I_NT1 + 3, 192, 64, 64);
  cvt(NT_IH2, I_NT2 + 2, 192, 128, 128);
  cvt(NT_HH2, I_NT2 + 3, 192, 64, 64);

  // Fused edge message passing (oneD then twoD)
  auto launch_sd = [&](int E, int ei, int ea, int xsrc, int hh, size_t wo,
                       int pb, float* Mo) {
    k_sd_edge<<<dim3((E + 127) / 128), 256, 0, stream>>>(
        E, (const int*)d_in[ei], (const float*)d_in[ea],
        (const float*)d_in[xsrc], (const float*)d_in[hh], WB + wo + MP_W1,
        WB + wo + MP_W2, WB + wo + MP_G1, WB + wo + MP_BG, WB + wo + MP_P1,
        WB + wo + MP_P2, (const float*)d_in[pb + 1], (const float*)d_in[pb + 3],
        (const float*)d_in[pb + 5], (const float*)d_in[pb + 7],
        (const float*)d_in[pb + 9], (const float*)d_in[pb + 11],
        (const float*)d_in[pb + 13], Mo);
  };
  launch_sd(BE1, I_EI1, I_EA1, I_XS1, I_H1, 0, I_MP1, M1);
  launch_sd(BE2, I_EI2, I_EA2, I_XS2, I_H2, MP_TOT, I_MP2, M2);

  // GAT projections
  k_lin64<<<dim3((int)((BN1 + 127) / 128)), 256, 0, stream>>>(
      (int)BN1, (const float*)d_in[I_H1], WB + G_LL,
      (const float*)d_in[I_GAT + 1], XL);
  k_lin64<<<dim3((int)((BN2 + 127) / 128)), 256, 0, stream>>>(
      (int)BN2, (const float*)d_in[I_H2], WB + G_LR,
      (const float*)d_in[I_GAT + 3], XR);

  // GAT edge softmax + scatter
  dim3 eg((BE12 + 255) / 256);
  k_gat_logit<<<eg, 256, 0, stream>>>(BE12, (const int*)d_in[I_EIX], XL, XR,
                                      (const float*)d_in[I_GAT + 4], LG, MAXB);
  k_gat_den<<<eg, 256, 0, stream>>>(BE12, (const int*)d_in[I_EIX], MAXB, LG,
                                    DEN);
  k_gat_scatter<<<eg, 256, 0, stream>>>(BE12, (const int*)d_in[I_EIX], XL, LG,
                                        DEN, GACC);

  // GAT node tail (residual + FFN) accumulated into M2
  k_gat_node<<<dim3((int)((BN2 + 127) / 128)), 256, 0, stream>>>(
      (int)BN2, GACC, (const float*)d_in[I_H2], WB + G_RES,
      (const float*)d_in[I_GAT + 6], WB + G_F1, (const float*)d_in[I_GAT + 8],
      WB + G_F2, (const float*)d_in[I_GAT + 10], M2);

  // Per-type node updates -> d_out
  auto launch_nt = [&](long N, float* Mo, int xd, int hh, size_t wih,
                       size_t whh, int pb, float* outp) {
    k_node_update<<<dim3((int)((N + 127) / 128)), 256, 0, stream>>>(
        (int)N, Mo, (const float*)d_in[xd], (const float*)d_in[hh], WB + wih,
        WB + whh, (const float*)d_in[pb + 4], (const float*)d_in[pb + 5],
        (const float*)d_in[pb + 0], (const float*)d_in[pb + 1],
        (const float*)d_in[pb + 6], (const float*)d_in[pb + 7],
        (const float*)d_in[pb + 8], (const float*)d_in[pb + 9],
        (const float*)d_in[pb + 10], (const float*)d_in[pb + 11], outp);
  };
  launch_nt(BN1, M1, I_XD1, I_H1, NT_IH1, NT_HH1, I_NT1, (float*)d_out);
  launch_nt(BN2, M2, I_XD2, I_H2, NT_IH2, NT_HH2, I_NT2,
            (float*)d_out + (size_t)BN1 * 64);
}
